// ShrinkingLayer_39685497815964
// MI455X (gfx1250) — matrix operand order repaired and checked by reading, not compiled, not run on hardware
//
#include <hip/hip_runtime.h>
#include <hip/hip_bf16.h>

// ---------------------------------------------------------------------------
// ShrinkingLayer for MI455X (gfx1250, wave32, WMMA).
//
// Key algebraic identity: edges are the complete graph (with self loops)
// within each cluster, so scatter-mean over edges reduces to:
//   aggr1[i,p] = (clusterMean - x_i) @ W_F[:,p,:] @ x_i + b_F[p,:]·x_i
// -> no edge traversal. Main compute = per-node bilinear forms, recast as
// GEMM with K = 272 (outer product d⊗x / x⊗x flattened + bias row), done
// with v_wmma_f32_16x16x32_f16 (f32 accumulate), 16 nodes per wave.
// ---------------------------------------------------------------------------

#define NB   8
#define II   2048
#define CC   16
#define CP   32
#define KK   64
#define NI   (NB * II)      // 16384
#define NSEG (NB * KK)      // 512

typedef __attribute__((ext_vector_type(16))) _Float16 v16h;
typedef __attribute__((ext_vector_type(8)))  float    v8f;

// ---------------- kernel 0: zero the accumulator region --------------------
__global__ __launch_bounds__(256) void zero_kernel(float* __restrict__ p, int n) {
    int i = blockIdx.x * 256 + threadIdx.x;
    if (i < n) p[i] = 0.0f;
}

// ---------------- kernel 1: SelfCorrelation + cluster/batch sums -----------
// x' = lr * x * (x @ W_mlp + b_mlp) + x ;  accumulate cluster sums + counts
// and per-batch sums (for Z1).
__global__ __launch_bounds__(256)
void selfcorr_kernel(const float* __restrict__ x_in, const int* __restrict__ cluster,
                     const float* __restrict__ lr_p,
                     const float* __restrict__ W_mlp, const float* __restrict__ b_mlp,
                     float* __restrict__ xw, float* __restrict__ csum,
                     float* __restrict__ ccnt, float* __restrict__ xsum) {
    __shared__ float Wm[256];
    __shared__ float bm[16];
    int tid = threadIdx.x;
    Wm[tid] = W_mlp[tid];
    if (tid < 16) bm[tid] = b_mlp[tid];
    __syncthreads();

    int i = blockIdx.x * 256 + tid;          // node id, NI total
    float lr = lr_p[0];
    float x[16], xo[16];
#pragma unroll
    for (int c = 0; c < 16; ++c) x[c] = x_in[i * 16 + c];
#pragma unroll
    for (int c = 0; c < 16; ++c) {
        float s = bm[c];
#pragma unroll
        for (int k = 0; k < 16; ++k) s += x[k] * Wm[k * 16 + c];
        xo[c] = lr * x[c] * s + x[c];
    }
    int cid = cluster[i];
    int n   = i >> 11;
#pragma unroll
    for (int c = 0; c < 16; ++c) {
        xw[i * 16 + c] = xo[c];
        atomicAdd(&csum[cid * 16 + c], xo[c]);
        atomicAdd(&xsum[n * 16 + c], xo[c]);
    }
    atomicAdd(&ccnt[cid], 1.0f);
}

// ---------------- kernel 2: pack B operand fragments -----------------------
// B' for tile (uv,pt): K rows 0..255 = W_{F or W}[k, p*16+c] (kc = k*16+c),
// K step 8 rows 0..15 = b_{F or W}[p*16+c]. Stored per lane as the
// v_wmma 16x16x32 f16 B operand: lane = col (p), halves e -> K = hi*16 + e (contiguous, same as A).
__global__ __launch_bounds__(256)
void bpack_kernel(const float* __restrict__ W_F, const float* __restrict__ b_F,
                  const float* __restrict__ W_W, const float* __restrict__ b_W,
                  _Float16* __restrict__ Bfrag) {
    int idx = blockIdx.x * 256 + threadIdx.x;   // 4 tiles * 9 steps * 32 lanes * 16
    if (idx >= 4 * 9 * 32 * 16) return;
    int e    = idx & 15;
    int lane = (idx >> 4) & 31;
    int s    = (idx >> 9) % 9;
    int tile = idx / (9 * 512);
    int hi = lane >> 4, nn = lane & 15;
    int uv = tile >> 1, pt = tile & 1;
    int p  = pt * 16 + nn;
    int kis = hi * 16 + e;                      // B fragment K mapping
    float val;
    if (s < 8) {
        int K = s * 32 + kis;
        int k = K >> 4, c = K & 15;
        val = (uv ? W_W : W_F)[k * 512 + p * 16 + c];
    } else {
        val = (kis < 16) ? (uv ? b_W : b_F)[p * 16 + kis] : 0.0f;
    }
    Bfrag[idx] = (_Float16)val;
}

// ---------------- kernel 3: main WMMA conv + fused epilogue ----------------
__global__ __launch_bounds__(256)
void conv_wmma_kernel(const float* __restrict__ xw, const int* __restrict__ cluster,
                      const float* __restrict__ csum, const float* __restrict__ ccnt,
                      const v16h* __restrict__ Bfrag,
                      const float* __restrict__ W_M, const float* __restrict__ b_M,
                      const float* __restrict__ W_B, const float* __restrict__ b_B,
                      float* __restrict__ conv, float* __restrict__ convsum) {
    __shared__ float Umat[8][16][33];
    __shared__ float Vmat[8][16][33];
    __shared__ float gbuf[8][2][16];

    const int w    = threadIdx.x >> 5;
    const int lane = threadIdx.x & 31;
    const int r    = lane & 15;
    const int hi   = lane >> 4;
    const int node = (blockIdx.x * 8 + w) * 16 + r;

    // Per-row operands: x row and d = clusterMean - x
    float x[16], dv[16];
#pragma unroll
    for (int c = 0; c < 16; ++c) x[c] = xw[node * 16 + c];
    int   cid = cluster[node];
    float inv = 1.0f / ccnt[cid];
#pragma unroll
    for (int c = 0; c < 16; ++c) dv[c] = csum[cid * 16 + c] * inv - x[c];

    // D = A x B accumulation: uv=0 -> U (aggr1, W_F path), uv=1 -> V (W_W path)
#pragma unroll
    for (int uv = 0; uv < 2; ++uv) {
#pragma unroll
        for (int pt = 0; pt < 2; ++pt) {
            v8f acc = {0.f, 0.f, 0.f, 0.f, 0.f, 0.f, 0.f, 0.f};
#pragma unroll
            for (int s = 0; s < 9; ++s) {
                v16h a;
#pragma unroll
                for (int e = 0; e < 16; ++e) {
                    // A fragment K mapping (contiguous: K = hi*16 + e, matches Bfrag)
                    int kis = hi * 16 + e;
                    float val;
                    if (s < 8) {
                        int K = s * 32 + kis;
                        int k = K >> 4, c = K & 15;
                        val = (uv ? x[k] : dv[k]) * x[c];   // outer product row
                    } else {
                        val = (kis < 16) ? x[kis] : 0.0f;   // bias row
                    }
                    a[e] = (_Float16)val;
                }
                v16h b = Bfrag[((uv * 2 + pt) * 9 + s) * 32 + lane];
                acc = __builtin_amdgcn_wmma_f32_16x16x32_f16(
                          false, a, false, b, (short)0, acc, false, false);
            }
            // D layout: lane holds column N = lane%16, rows M = v + 8*hi
            float* dst = uv ? &Vmat[w][0][0] : &Umat[w][0][0];
            int p = pt * 16 + r;
#pragma unroll
            for (int v = 0; v < 8; ++v) dst[(v + 8 * hi) * 33 + p] = acc[v];
        }
    }
    __syncthreads();

    // ---- epilogue: gate by W_M, add W_W term, +W_B residual, relu ----
    float part = 0.0f;
#pragma unroll
    for (int t = 0; t < 16; ++t) part += Umat[w][r][hi * 16 + t] * W_M[hi * 16 + t];
    gbuf[w][hi][r] = part;
    __syncthreads();
    float g = gbuf[w][0][r] + gbuf[w][1][r] + b_M[0];

    float a3[16];
#pragma unroll
    for (int t = 0; t < 16; ++t) {
        int p = hi * 16 + t;
        a3[t] = Umat[w][r][p] * g + Vmat[w][r][p];
    }
    __syncthreads();
#pragma unroll
    for (int t = 0; t < 16; ++t) Umat[w][r][hi * 16 + t] = a3[t];
    __syncthreads();

    int n = node >> 11;
#pragma unroll
    for (int t = 0; t < 16; ++t) {
        int p = hi * 16 + t;
        float s = a3[t] + b_B[p];
        for (int q = 0; q < 32; ++q) s += Umat[w][r][q] * W_B[q * 32 + p];
        float cv = fmaxf(s, 0.0f);
        conv[node * 32 + p] = cv;
        atomicAdd(&convsum[n * 32 + p], cv);
    }
}

// ---------------- kernel 4: LAFA mixing weights (softmax) ------------------
__global__ __launch_bounds__(256)
void lafa_kernel(const float* __restrict__ xsum, const float* __restrict__ convsum,
                 const float* __restrict__ W_m1, const float* __restrict__ b_m1,
                 const float* __restrict__ W_m2, const float* __restrict__ b_m2,
                 float* __restrict__ Mw) {
    int tid = threadIdx.x;        // 256 = 8 batches x 32 channels
    int n = tid >> 5, p = tid & 31;
    const float invI = 1.0f / (float)II;
    float z1 = b_m1[p];
    for (int q = 0; q < 16; ++q) z1 += (xsum[n * 16 + q] * invI) * W_m1[q * 32 + p];
    float z2 = b_m2[p];
    for (int q = 0; q < 32; ++q) z2 += (convsum[n * 32 + q] * invI) * W_m2[q * 32 + p];
    float m  = fmaxf(z1, z2);
    float e1 = __expf(z1 - m), e2 = __expf(z2 - m);
    float iv = 1.0f / (e1 + e2);
    Mw[n * 32 + p]       = e1 * iv;
    Mw[256 + n * 32 + p] = e2 * iv;
}

// ---------------- kernel 5: fuse + cluster max-pool (uint-keyed) -----------
__global__ __launch_bounds__(256)
void fuse_pool_kernel(const float* __restrict__ xw, const float* __restrict__ conv,
                      const float* __restrict__ Mw, const int* __restrict__ cluster,
                      unsigned int* __restrict__ pooled) {
    int idx = blockIdx.x * 256 + threadIdx.x;   // NI * 32
    if (idx >= NI * 32) return;
    int i = idx >> 5, p = idx & 31;
    int n = i >> 11;
    float xp = (p < 16) ? xw[i * 16 + p] : 0.0f;
    float f  = Mw[n * 32 + p] * xp + Mw[256 + n * 32 + p] * conv[idx];
    unsigned int b   = __float_as_uint(f);
    unsigned int key = (b & 0x80000000u) ? ~b : (b | 0x80000000u);  // monotone map
    atomicMax(&pooled[cluster[i] * 32 + p], key);
}

// ---------------- kernel 6: unmap keys into d_out --------------------------
__global__ __launch_bounds__(256)
void unmap_kernel(const unsigned int* __restrict__ pooled, float* __restrict__ out) {
    int idx = blockIdx.x * 256 + threadIdx.x;   // NSEG * 32
    if (idx >= NSEG * 32) return;
    unsigned int key = pooled[idx];
    unsigned int b   = (key & 0x80000000u) ? (key & 0x7FFFFFFFu) : ~key;
    out[idx] = __uint_as_float(b);
}

// ---------------------------------------------------------------------------
extern "C" void kernel_launch(void* const* d_in, const int* in_sizes, int n_in,
                              void* d_out, int out_size, void* d_ws, size_t ws_size,
                              hipStream_t stream) {
    (void)in_sizes; (void)n_in; (void)out_size; (void)ws_size;
    const float* x_in    = (const float*)d_in[0];
    const int*   cluster = (const int*)d_in[1];
    /* d_in[2] = edge_index: unused — complete-graph-per-cluster identity */
    const float* lr      = (const float*)d_in[3];
    const float* W_mlp   = (const float*)d_in[4];
    const float* b_mlp   = (const float*)d_in[5];
    const float* W_F     = (const float*)d_in[6];
    const float* b_F     = (const float*)d_in[7];
    const float* W_W     = (const float*)d_in[8];
    const float* b_W     = (const float*)d_in[9];
    const float* W_M     = (const float*)d_in[10];
    const float* b_M     = (const float*)d_in[11];
    const float* W_B     = (const float*)d_in[12];
    const float* b_B     = (const float*)d_in[13];
    const float* W_m1    = (const float*)d_in[14];
    const float* b_m1    = (const float*)d_in[15];
    const float* W_m2    = (const float*)d_in[16];
    const float* b_m2    = (const float*)d_in[17];

    // workspace layout (all offsets 256B-aligned; zeroed region contiguous)
    char* base = (char*)d_ws;
    float*        csum    = (float*)(base + 0);            //  512*16
    float*        ccnt    = (float*)(base + 32768);        //  512
    float*        xsum    = (float*)(base + 34816);        //  8*16
    float*        convsum = (float*)(base + 35328);        //  8*32
    unsigned int* pooled  = (unsigned int*)(base + 36352); //  512*32
    const int ZERO_ELEMS  = 101888 / 4;                    //  25472 floats
    float*        xw      = (float*)(base + 101888);       //  NI*16
    float*        conv    = (float*)(base + 101888 + 1048576);           // NI*32
    float*        Mw      = (float*)(base + 101888 + 1048576 + 2097152); // 2*8*32
    _Float16*     Bfrag   = (_Float16*)(base + 101888 + 1048576 + 2097152 + 2048);

    zero_kernel<<<(ZERO_ELEMS + 255) / 256, 256, 0, stream>>>((float*)base, ZERO_ELEMS);
    selfcorr_kernel<<<NI / 256, 256, 0, stream>>>(x_in, cluster, lr, W_mlp, b_mlp,
                                                  xw, csum, ccnt, xsum);
    bpack_kernel<<<(4 * 9 * 32 * 16 + 255) / 256, 256, 0, stream>>>(W_F, b_F, W_W, b_W, Bfrag);
    conv_wmma_kernel<<<NI / (8 * 16), 256, 0, stream>>>(xw, cluster, csum, ccnt,
                                                        (const v16h*)Bfrag,
                                                        W_M, b_M, W_B, b_B,
                                                        conv, convsum);
    lafa_kernel<<<1, 256, 0, stream>>>(xsum, convsum, W_m1, b_m1, W_m2, b_m2, Mw);
    fuse_pool_kernel<<<(NI * 32) / 256, 256, 0, stream>>>(xw, conv, Mw, cluster, pooled);
    unmap_kernel<<<(NSEG * 32 + 255) / 256, 256, 0, stream>>>(pooled, (float*)d_out);
}
